// LossFunction_103079215159
// MI455X (gfx1250) — compile-verified, matrix operand order checked by reading, not run on hardware
//
#include <hip/hip_runtime.h>
#include <hip/hip_bf16.h>
#include <stdint.h>

// ---------------- problem constants ----------------
#define B_ROWS 8192
#define DDIM   256
#define EPSV   1e-6f
#define NCHUNK 4
#define COLS_PER_CHUNK (B_ROWS / NCHUNK)        // 2048
#define TILES_PER_CHUNK (COLS_PER_CHUNK / 16)   // 128
#define WAVES_PER_BLOCK2 8                      // 128 rows per block

typedef __attribute__((ext_vector_type(16))) _Float16 v16h;
typedef __attribute__((ext_vector_type(8)))  _Float16 h8;
typedef __attribute__((ext_vector_type(8)))  float    v8f;
typedef __attribute__((ext_vector_type(4)))  unsigned int v4u;
typedef __attribute__((ext_vector_type(8)))  int v8i;
typedef __attribute__((ext_vector_type(4)))  int v4i;

// ---------------- helpers ----------------
__device__ __forceinline__ float wred32(float v) {
#pragma unroll
  for (int m = 16; m >= 1; m >>= 1) v += __shfl_xor(v, m, 32);
  return v;
}

// 16x32 f16 WMMA fragment per ISA layout (works for global or LDS pointers):
// lanes 0-15 take halves [k0, k0+8) and [k0+16, k0+24); lanes 16-31 shift +8.
__device__ __forceinline__ v16h frag_ld(const _Float16* p) {
  h8 lo = *(const h8*)(p);
  h8 hi = *(const h8*)(p + 16);
  return __builtin_shufflevector(lo, hi, 0,1,2,3,4,5,6,7,8,9,10,11,12,13,14,15);
}

struct Top5 { float v[5]; int ix[5]; };

__device__ __forceinline__ void top5_init(Top5& t) {
#pragma unroll
  for (int k = 0; k < 5; ++k) { t.v[k] = __builtin_inff(); t.ix[k] = 0; }
}

__device__ __forceinline__ void top5_insert(Top5& t, float kv, int kj) {
  if (kv < t.v[4]) {
    bool c0 = kv < t.v[0], c1 = kv < t.v[1], c2 = kv < t.v[2], c3 = kv < t.v[3];
    t.v[4] = c3 ? t.v[3] : kv;                 t.ix[4] = c3 ? t.ix[3] : kj;
    t.v[3] = c2 ? t.v[2] : (c3 ? kv : t.v[3]); t.ix[3] = c2 ? t.ix[2] : (c3 ? kj : t.ix[3]);
    t.v[2] = c1 ? t.v[1] : (c2 ? kv : t.v[2]); t.ix[2] = c1 ? t.ix[1] : (c2 ? kj : t.ix[2]);
    t.v[1] = c0 ? t.v[0] : (c1 ? kv : t.v[1]); t.ix[1] = c0 ? t.ix[0] : (c1 ? kj : t.ix[1]);
    t.v[0] = c0 ? kv : t.v[0];                 t.ix[0] = c0 ? kj : t.ix[0];
  }
}

// TDM: load one 16-row x 256-half row-major tile (512B/row) from global into LDS.
// D# packing per CDNA5 ISA 8.3-8.6: group0 {count=1, lds_addr, global_addr, type=2},
// group1 {data_size=2B, tensor 256x8192, tile 256x16, stride0=256}. 2D => groups 2/3 zero.
__device__ __forceinline__ void tdm_load_tile(const _Float16* gsrc, unsigned int ldsOff) {
  uint64_t ga = (uint64_t)(uintptr_t)gsrc;
  v4u g0;
  g0[0] = 1u;                                   // count=1 (valid descriptor)
  g0[1] = ldsOff;                               // lds_addr (bytes)
  g0[2] = (unsigned int)ga;                     // global_addr[31:0]
  g0[3] = (unsigned int)(ga >> 32) | (2u << 30);// global_addr[56:32] | type=2
  v8i g1;
  g1[0] = (int)(1u << 16);                      // data_size=1 (2 bytes)
  g1[1] = (int)(256u << 16);                    // tensor_dim0 = 256
  g1[2] = (int)(8192u << 16);                   // tensor_dim1 = 8192 (low16)
  g1[3] = (int)(256u << 16);                    // tile_dim0 = 256
  g1[4] = 16;                                   // tile_dim1 = 16
  g1[5] = 256;                                  // tensor_dim0_stride = 256
  g1[6] = 0;
  g1[7] = 0;
  v4i z4 = {0, 0, 0, 0};
  v8i z8 = {0, 0, 0, 0, 0, 0, 0, 0};
  __builtin_amdgcn_tensor_load_to_lds(g0, g1, z4, z4, z8, 0);
}

__device__ __forceinline__ unsigned int lds_off(const void* p) {
  // flat LDS address = {shared aperture, lds offset}; low 32 bits are the LDS offset
  return (unsigned int)(uintptr_t)p;
}

// JAX threefry2x32 (20 rounds), bit-exact
__device__ __forceinline__ void threefry2x32(uint32_t k0, uint32_t k1,
                                             uint32_t x0, uint32_t x1,
                                             uint32_t& o0, uint32_t& o1) {
  uint32_t ks[3] = { k0, k1, k0 ^ k1 ^ 0x1BD11BDAu };
  const int R0[4] = {13, 15, 26, 6}, R1[4] = {17, 29, 16, 24};
  x0 += ks[0]; x1 += ks[1];
#pragma unroll
  for (int i = 0; i < 5; ++i) {
    const int* rot = (i & 1) ? R1 : R0;
#pragma unroll
    for (int j = 0; j < 4; ++j) {
      x0 += x1;
      x1 = (x1 << rot[j]) | (x1 >> (32 - rot[j]));
      x1 ^= x0;
    }
    x0 += ks[(i + 1) % 3];
    x1 += ks[(i + 2) % 3] + (uint32_t)(i + 1);
  }
  o0 = x0; o1 = x1;
}

// ---------------- kernel 1: row normalize + split-f16 encode ----------------
__global__ __launch_bounds__(256)
void k_normalize(const float* __restrict__ x,
                 _Float16* __restrict__ ah, _Float16* __restrict__ al,
                 _Float16* __restrict__ ph, _Float16* __restrict__ pl,
                 float* __restrict__ rp) {
  int wave = (int)((blockIdx.x * blockDim.x + threadIdx.x) >> 5);
  int lane = threadIdx.x & 31;
  int row = wave >> 1, which = wave & 1;
  if (row >= B_ROWS) return;

  const float* src = x + (size_t)row * (2 * DDIM) + which * DDIM + lane * 8;
  const float4* s4 = (const float4*)src;
  float4 f0 = s4[0], f1 = s4[1];
  float v[8] = { f0.x, f0.y, f0.z, f0.w, f1.x, f1.y, f1.z, f1.w };

  float ss = 0.f;
#pragma unroll
  for (int e = 0; e < 8; ++e) ss += v[e] * v[e];
  ss = wred32(ss);
  float inv = 1.0f / fmaxf(sqrtf(ss), 1e-12f);

  float rs = 0.f;
  h8 hv, lv;
#pragma unroll
  for (int e = 0; e < 8; ++e) {
    float nv = v[e] * inv;
    rs += nv;
    _Float16 h = (_Float16)nv;
    hv[e] = h;
    lv[e] = (_Float16)(nv - (float)h);
  }
  rs = wred32(rs);

  _Float16* dh = which ? ph : ah;
  _Float16* dl = which ? pl : al;
  *(h8*)(dh + (size_t)row * DDIM + lane * 8) = hv;
  *(h8*)(dl + (size_t)row * DDIM + lane * 8) = lv;
  if (which == 1 && lane == 0) rp[row] = rs;
}

// ---------- kernel 2: TDM-staged split-f16 WMMA GEMM + streaming per-row top-5 ----------
__global__ __launch_bounds__(256)
void k_gemm_topk(const _Float16* __restrict__ ah, const _Float16* __restrict__ al,
                 const _Float16* __restrict__ ph, const _Float16* __restrict__ pl,
                 const float* __restrict__ rp,
                 float* __restrict__ partVal, int* __restrict__ partIdx) {
  __shared__ _Float16 sBh[2][16 * DDIM];   // 2 x 8KB, hi halves of B tile
  __shared__ _Float16 sBl[2][16 * DDIM];   // 2 x 8KB, lo halves of B tile

  int w    = threadIdx.x >> 5;
  int lane = threadIdx.x & 31;
  int lrow = lane & 15;
  int hi   = lane >> 4;
  int sel  = hi << 3;                       // 0 or 8 (K-half select)
  int rowBase  = ((int)blockIdx.x * WAVES_PER_BLOCK2 + w) * 16;
  int chunk    = (int)blockIdx.y;
  int colStart = chunk * COLS_PER_CHUNK;

  // A fragments (hi + lo split) held in registers for the whole column sweep
  v16h aH[8], aL[8];
#pragma unroll
  for (int kt = 0; kt < 8; ++kt) {
    const size_t ao = (size_t)(rowBase + lrow) * DDIM + kt * 32 + sel;
    aH[kt] = frag_ld(ah + ao);
    aL[kt] = frag_ld(al + ao);
  }

  Top5 tk[8];
#pragma unroll
  for (int r = 0; r < 8; ++r) top5_init(tk[r]);

  // prologue: TDM-stage tile 0
  if (w == 0) {
    tdm_load_tile(ph + (size_t)colStart * DDIM, lds_off(&sBh[0][0]));
    tdm_load_tile(pl + (size_t)colStart * DDIM, lds_off(&sBl[0][0]));
  }

  for (int t = 0; t < TILES_PER_CHUNK; ++t) {
    int cur = t & 1;
    if (w == 0) __builtin_amdgcn_s_wait_tensorcnt((short)0);
    __syncthreads();                         // tile t visible to all waves
    if (w == 0 && t + 1 < TILES_PER_CHUNK) { // prefetch tile t+1 into other buffer
      size_t off = (size_t)(colStart + (t + 1) * 16) * DDIM;
      tdm_load_tile(ph + off, lds_off(&sBh[cur ^ 1][0]));
      tdm_load_tile(pl + off, lds_off(&sBl[cur ^ 1][0]));
    }

    int colBase = colStart + t * 16;
    int n = colBase + lrow;                  // this lane's global column
    float keyBase = -2.0f * EPSV * rp[n];

    v8f acc = {0.f, 0.f, 0.f, 0.f, 0.f, 0.f, 0.f, 0.f};
#pragma unroll
    for (int kt = 0; kt < 8; ++kt) {
      v16h bh = frag_ld(&sBh[cur][lrow * DDIM + kt * 32 + sel]);
      v16h bl = frag_ld(&sBl[cur][lrow * DDIM + kt * 32 + sel]);
      acc = __builtin_amdgcn_wmma_f32_16x16x32_f16(false, aH[kt], false, bh,
                                                   (short)0, acc, false, false);
      acc = __builtin_amdgcn_wmma_f32_16x16x32_f16(false, aH[kt], false, bl,
                                                   (short)0, acc, false, false);
      acc = __builtin_amdgcn_wmma_f32_16x16x32_f16(false, aL[kt], false, bh,
                                                   (short)0, acc, false, false);
    }

    // key = -2*ip - 2*eps*rp[n] : same per-row ordering as d2
#pragma unroll
    for (int r = 0; r < 8; ++r) {
      int m = rowBase + r + 8 * hi;
      float kv = fmaf(-2.0f, acc[r], keyBase);
      kv = (m == n) ? __builtin_inff() : kv; // exclude diagonal
      top5_insert(tk[r], kv, n);
    }
    __syncthreads();                         // done reading buf before its reuse
  }

  // cross-lane merge: xor-butterfly within each 16-lane half (snapshot then insert)
#pragma unroll
  for (int mask = 1; mask <= 8; mask <<= 1) {
#pragma unroll
    for (int r = 0; r < 8; ++r) {
      float ov[5]; int oi[5];
#pragma unroll
      for (int k = 0; k < 5; ++k) {
        ov[k] = __shfl_xor(tk[r].v[k], mask, 32);
        oi[k] = __shfl_xor(tk[r].ix[k], mask, 32);
      }
#pragma unroll
      for (int k = 0; k < 5; ++k) top5_insert(tk[r], ov[k], oi[k]);
    }
  }

  // lane (hi*16 + r) writes row rowBase + r + 8*hi
#pragma unroll
  for (int r = 0; r < 8; ++r) {
    if (lrow == r) {
      int grow = rowBase + r + 8 * hi;
#pragma unroll
      for (int k = 0; k < 5; ++k) {
        partVal[((size_t)grow * NCHUNK + chunk) * 5 + k] = tk[r].v[k];
        partIdx[((size_t)grow * NCHUNK + chunk) * 5 + k] = tk[r].ix[k];
      }
    }
  }
}

// ---------------- kernel 3: merge chunk partials -> global top-5 indices ----------------
__global__ __launch_bounds__(256)
void k_merge(const float* __restrict__ partVal, const int* __restrict__ partIdx,
             int* __restrict__ topidx) {
  int row = (int)(blockIdx.x * blockDim.x + threadIdx.x);
  if (row >= B_ROWS) return;
  Top5 g; top5_init(g);
  for (int c = 0; c < NCHUNK; ++c) {
#pragma unroll
    for (int k = 0; k < 5; ++k)
      top5_insert(g, partVal[((size_t)row * NCHUNK + c) * 5 + k],
                     partIdx[((size_t)row * NCHUNK + c) * 5 + k]);
  }
#pragma unroll
  for (int k = 0; k < 5; ++k) topidx[row * 5 + k] = g.ix[k];
}

// ---------------- kernel 4: threefry selection + triplet loss reduction ----------------
__global__ __launch_bounds__(256)
void k_loss(const float* __restrict__ x, const int* __restrict__ topidx,
            float* __restrict__ out) {
  int wave = (int)((blockIdx.x * blockDim.x + threadIdx.x) >> 5);
  int lane = threadIdx.x & 31;
  if (wave >= B_ROWS) return;
  int i = wave;

  // split(key(1)) -> k1=(a0,a1), k2=(b0,b1); JAX halves pairing of counts 0..3
  uint32_t a0, b0, a1, b1;
  threefry2x32(0u, 1u, 0u, 2u, a0, b0);
  threefry2x32(0u, 1u, 1u, 3u, a1, b1);

  uint32_t i0 = (i < 4096) ? (uint32_t)i : (uint32_t)(i - 4096);
  uint32_t u0, u1; threefry2x32(a0, a1, i0, i0 + 4096u, u0, u1);
  uint32_t ubits = (i < 4096) ? u0 : u1;
  float uf = __uint_as_float(0x3F800000u | (ubits >> 9)) - 1.0f;
  bool coin = uf < 0.5f;

  uint32_t r0, r1; threefry2x32(b0, b1, i0, i0 + 4096u, r0, r1);
  uint32_t rbits = (i < 4096) ? r0 : r1;
  uint32_t rank = coin ? 0u : (((rbits >> 16) + (rbits & 0xFFFFu)) % 5u);
  int negidx = topidx[i * 5 + (int)rank];

  const float* pa = x + (size_t)i * (2 * DDIM) + lane * 8;
  const float* pp = pa + DDIM;
  const float* pn = x + (size_t)negidx * (2 * DDIM) + DDIM + lane * 8;
  float4 A0 = ((const float4*)pa)[0], A1 = ((const float4*)pa)[1];
  float4 P0 = ((const float4*)pp)[0], P1 = ((const float4*)pp)[1];
  float4 N0 = ((const float4*)pn)[0], N1 = ((const float4*)pn)[1];
  float va[8] = {A0.x, A0.y, A0.z, A0.w, A1.x, A1.y, A1.z, A1.w};
  float vp[8] = {P0.x, P0.y, P0.z, P0.w, P1.x, P1.y, P1.z, P1.w};
  float vn[8] = {N0.x, N0.y, N0.z, N0.w, N1.x, N1.y, N1.z, N1.w};

  float sa = 0.f, sp = 0.f, sn = 0.f;
#pragma unroll
  for (int e = 0; e < 8; ++e) { sa += va[e]*va[e]; sp += vp[e]*vp[e]; sn += vn[e]*vn[e]; }
  sa = wred32(sa); sp = wred32(sp); sn = wred32(sn);
  float ia  = 1.0f / fmaxf(sqrtf(sa), 1e-12f);
  float ipv = 1.0f / fmaxf(sqrtf(sp), 1e-12f);
  float inv = 1.0f / fmaxf(sqrtf(sn), 1e-12f);

  float pd = 0.f, nd = 0.f;
#pragma unroll
  for (int e = 0; e < 8; ++e) {
    float aa = va[e] * ia;
    float d1 = aa - vp[e] * ipv + EPSV; pd += d1 * d1;
    float d2 = aa - vn[e] * inv + EPSV; nd += d2 * d2;
  }
  pd = wred32(pd); nd = wred32(nd);
  if (lane == 0) {
    float l = fmaxf(pd - nd, 0.0f);
    atomicAdd(out, l * (1.0f / (float)B_ROWS));
  }
}

// ---------------- launcher ----------------
extern "C" void kernel_launch(void* const* d_in, const int* in_sizes, int n_in,
                              void* d_out, int out_size, void* d_ws, size_t ws_size,
                              hipStream_t stream) {
  (void)in_sizes; (void)n_in; (void)ws_size;
  const float* x = (const float*)d_in[0];
  char* ws = (char*)d_ws;

  const size_t SZ_HALF = (size_t)B_ROWS * DDIM * sizeof(_Float16); // 4 MB
  _Float16* ah = (_Float16*)(ws);
  _Float16* al = (_Float16*)(ws + SZ_HALF);
  _Float16* ph = (_Float16*)(ws + 2 * SZ_HALF);
  _Float16* pl = (_Float16*)(ws + 3 * SZ_HALF);
  float* rp      = (float*)(ws + 4 * SZ_HALF);                       // 32 KB
  float* partVal = (float*)(ws + 4 * SZ_HALF + 32768);               // 640 KB
  int*   partIdx = (int*)  (ws + 4 * SZ_HALF + 32768 + 655360);      // 640 KB
  int*   topidx  = (int*)  (ws + 4 * SZ_HALF + 32768 + 2 * 655360);  // 160 KB

  // 1) normalize + split-f16 encode (2*8192 rows, one wave per row)
  k_normalize<<<2048, 256, 0, stream>>>(x, ah, al, ph, pl, rp);

  // 2) TDM-staged WMMA GEMM + streaming top-5 (4 column chunks)
  dim3 g2(B_ROWS / (16 * WAVES_PER_BLOCK2), NCHUNK);
  k_gemm_topk<<<g2, 256, 0, stream>>>(ah, al, ph, pl, rp, partVal, partIdx);

  // 3) merge chunk partials
  k_merge<<<B_ROWS / 256, 256, 0, stream>>>(partVal, partIdx, topidx);

  // 4) loss (zero the scalar accumulator first)
  (void)hipMemsetAsync(d_out, 0, sizeof(float) * (out_size > 0 ? out_size : 1), stream);
  k_loss<<<(B_ROWS * 32) / 256, 256, 0, stream>>>(x, topidx, (float*)d_out);
}
